// ZConditionedReadout_49950469652578
// MI455X (gfx1250) — matrix operand order repaired and checked by reading, not compiled, not run on hardware
//
#include <hip/hip_runtime.h>

// Problem constants (match reference)
#define N_ATOMS   4096
#define N_NEIGH   32
#define IN_F      128
#define HID       256
#define OUT_F     128
#define N_SPECIES 64

typedef __attribute__((ext_vector_type(16))) _Float16 v16h;
typedef __attribute__((ext_vector_type(8)))  float    v8f;
typedef __attribute__((ext_vector_type(2)))  __fp16   hf2;
typedef unsigned int u32x4 __attribute__((ext_vector_type(4)));
typedef int          i32x8 __attribute__((ext_vector_type(8)));
typedef int          i32x4 __attribute__((ext_vector_type(4)));

union FragU { v16h h; unsigned int u[8]; };

// Single-instruction packed f32x2 -> f16x2 (v_cvt_pk_rtz_f16_f32).
__device__ __forceinline__ unsigned int pack2(float a, float b) {
  hf2 p = __builtin_amdgcn_cvt_pkrtz(a, b);
  return __builtin_bit_cast(unsigned int, p);
}

// Padded LDS row strides (u32 units) -> bank-conflict-free fragment gathers.
// TDM reproduces these pitches in hardware via pad_enable/pad_interval/pad_amount.
#define XSTRIDE   65   // X:  32 rows x (128 f16 / 2 + 1)
#define W1STRIDE  65   // W1: 256 rows x (128 f16 / 2 + 1)
#define W2STRIDE  129  // W2: 128 rows x (256 f16 / 2 + 1)
#define HSTRIDE   129  // H:  32 rows x (256 f16 / 2 + 1), read as u32

// ---------------------------------------------------------------------------
// TDM: one instruction DMAs a whole [rows x row_elems_f16] f16 tile from
// global into LDS, inserting 1 dword of padding after each row
// (pad_interval_code: 5 -> every 64 dwords, 6 -> every 128 dwords).
// D# layout per CDNA5 ISA. 2D tile ->
// groups 2/3 (and the trailing v8i operand) are unused and zero-filled.
// ---------------------------------------------------------------------------
__device__ __forceinline__ void tdm_load_2d_f16(unsigned lds_addr,
                                                unsigned long long gaddr,
                                                unsigned rows,
                                                unsigned row_elems,
                                                unsigned pad_interval_code) {
  u32x4 g0;
  g0[0] = 1u;                                             // count=1, user D#
  g0[1] = lds_addr;                                       // lds_addr [63:32]
  g0[2] = (unsigned)(gaddr & 0xFFFFFFFFull);              // global_addr lo
  g0[3] = (unsigned)((gaddr >> 32) & 0x1FFFFFFull)        // global_addr hi
        | (2u << 30);                                     // type=2 ("image")
  i32x8 g1;
  g1[0] = (int)((1u << 16)                                // data_size = 2B
              | (1u << 20)                                // pad_enable
              | (pad_interval_code << 22)                 // dwords before pad
              | (0u << 25));                              // pad_amount = 1 dword
  g1[1] = (int)((row_elems & 0xFFFFu) << 16);             // tensor_dim0 [15:0]
  g1[2] = (int)(((row_elems >> 16) & 0xFFFFu)             // tensor_dim0 [31:16]
              | ((rows & 0xFFFFu) << 16));                // tensor_dim1 [15:0]
  g1[3] = (int)(((rows >> 16) & 0xFFFFu)                  // tensor_dim1 [31:16]
              | ((row_elems & 0xFFFFu) << 16));           // tile_dim0
  g1[4] = (int)(rows & 0xFFFFu);                          // tile_dim1, tile_dim2=0
  g1[5] = (int)row_elems;                                 // tensor_dim0_stride lo32
  g1[6] = 0;                                              // stride hi, dim1_stride lo
  g1[7] = 0;
  i32x4 z4 = {0, 0, 0, 0};
  i32x8 z8 = {0, 0, 0, 0, 0, 0, 0, 0};
  __builtin_amdgcn_tensor_load_to_lds(g0, g1, z4, z4, z8, 0);
}

// ---------------------------------------------------------------------------
// Prep: one-time f32 -> f16 conversion of all species weights into d_ws.
// W1h: [64][256][128] f16 at ws+0; W2h: [64][128][256] f16 at ws+4MB.
// ---------------------------------------------------------------------------
__global__ __launch_bounds__(256) void convert_weights_kernel(
    const float* __restrict__ W1, const float* __restrict__ W2,
    unsigned int* __restrict__ ws) {
  const int t = blockIdx.x * 256 + threadIdx.x;           // 524288 threads
  const int n2 = (N_SPECIES * HID * IN_F) / 2;            // float2 count per array
  #pragma unroll
  for (int i = 0; i < 2; ++i) {
    int e = t + i * (2048 * 256);
    float2 a = ((const float2*)W1)[e];
    ws[e] = pack2(a.x, a.y);
    float2 b = ((const float2*)W2)[e];
    ws[n2 + e] = pack2(b.x, b.y);
  }
}

// ---------------------------------------------------------------------------
// Main kernel. PRE=true: weights already f16 in ws, staged to LDS by the TDM.
// PRE=false: self-contained per-block f32->f16 conversion staging.
// ---------------------------------------------------------------------------
template <bool PRE>
__global__ __launch_bounds__(256) void zcond_readout_kernel(
    const float* __restrict__ features, const int* __restrict__ species,
    const float* __restrict__ W1g, const float* __restrict__ b1g,
    const float* __restrict__ W2g, const float* __restrict__ b2g,
    const unsigned int* __restrict__ W1h,   // f16-packed, PRE only
    const unsigned int* __restrict__ W2h,   // f16-packed, PRE only
    float* __restrict__ out) {
  __shared__ unsigned int Xlds[N_NEIGH * XSTRIDE];        // ~8.3 KB
  __shared__ unsigned int Wlds[HID * W1STRIDE];           // ~65  KB (W1 then W2)
  __shared__ unsigned int Hlds[N_NEIGH * HSTRIDE];        // ~16  KB

  const int atom = blockIdx.x;
  const int tid  = threadIdx.x;
  const int lane = tid & 31;
  const int wave = tid >> 5;
  const int col  = lane & 15;
  const int hsel = (lane >> 4) & 1;
  const int s    = species[atom];

  // ---- Stage X: 32x128 f32 -> f16 LDS ----
  {
    const float4* src = (const float4*)(features + (size_t)atom * N_NEIGH * IN_F);
    #pragma unroll
    for (int i = 0; i < 4; ++i) {
      int e   = tid + i * 256;
      int row = e >> 5, off = (e & 31) * 2;
      float4 v = src[e];
      Xlds[row * XSTRIDE + off]     = pack2(v.x, v.y);
      Xlds[row * XSTRIDE + off + 1] = pack2(v.z, v.w);
    }
  }
  // ---- Stage W1 into LDS ----
  if (PRE) {
    if (wave == 0) {
      tdm_load_2d_f16((unsigned)(size_t)&Wlds[0],
                      (unsigned long long)(size_t)W1h + (size_t)s * HID * IN_F * 2,
                      HID, IN_F, 5u /* pad every 64 dwords */);
      __builtin_amdgcn_s_wait_tensorcnt(0);
    }
  } else {
    const float4* src = (const float4*)(W1g + (size_t)s * HID * IN_F);
    #pragma unroll 8
    for (int i = 0; i < 32; ++i) {
      int e   = tid + i * 256;
      int row = e >> 5, off = (e & 31) * 2;
      float4 v = src[e];
      Wlds[row * W1STRIDE + off]     = pack2(v.x, v.y);
      Wlds[row * W1STRIDE + off + 1] = pack2(v.z, v.w);
    }
  }
  __syncthreads();

  // ---- Layer 1: H[32x256] = X[32x128] . W1^T  (+b1, SiLU) ----
  {
    const int mtile  = wave & 1;
    const int ntbase = wave >> 1;
    FragU a[4];
    {
      const unsigned int* P = Xlds + (mtile * 16 + col) * XSTRIDE + hsel * 4;
      #pragma unroll
      for (int ks = 0; ks < 4; ++ks)
        #pragma unroll
        for (int v = 0; v < 4; ++v) {
          a[ks].u[v]     = P[ks * 16 + v];
          a[ks].u[4 + v] = P[ks * 16 + 8 + v];
        }
    }
    #pragma unroll
    for (int t = 0; t < 4; ++t) {
      const int n = (ntbase + 4 * t) * 16 + col;
      const unsigned int* Q = Wlds + n * W1STRIDE + hsel * 4;
      v8f acc = {};
      #pragma unroll
      for (int ks = 0; ks < 4; ++ks) {
        FragU b;
        #pragma unroll
        for (int v = 0; v < 8; ++v) b.u[v] = Q[ks * 16 + (v & 3) + ((v >> 2) << 3)];
        acc = __builtin_amdgcn_wmma_f32_16x16x32_f16(
                  false, a[ks].h, false, b.h, (short)0, acc, false, false);
      }
      const float bias = b1g[s * HID + n];
      unsigned short* Hh = (unsigned short*)Hlds;
      #pragma unroll
      for (int r = 0; r < 8; ++r) {
        float x = acc[r] + bias;
        float y = x / (1.0f + __expf(-x));               // SiLU
        const int m = mtile * 16 + hsel * 8 + r;
        Hh[m * (2 * HSTRIDE) + n] =
            __builtin_bit_cast(unsigned short, (_Float16)y);
      }
    }
  }
  __syncthreads();  // W1 consumed, H complete

  // ---- Stage W2 into LDS (reuse buffer) ----
  if (PRE) {
    if (wave == 0) {
      tdm_load_2d_f16((unsigned)(size_t)&Wlds[0],
                      (unsigned long long)(size_t)W2h + (size_t)s * OUT_F * HID * 2,
                      OUT_F, HID, 6u /* pad every 128 dwords */);
      __builtin_amdgcn_s_wait_tensorcnt(0);
    }
  } else {
    const float4* src = (const float4*)(W2g + (size_t)s * OUT_F * HID);
    #pragma unroll 8
    for (int i = 0; i < 32; ++i) {
      int e   = tid + i * 256;
      int row = e >> 6, off = (e & 63) * 2;
      float4 v = src[e];
      Wlds[row * W2STRIDE + off]     = pack2(v.x, v.y);
      Wlds[row * W2STRIDE + off + 1] = pack2(v.z, v.w);
    }
  }
  __syncthreads();

  // ---- Layer 2: Out[32x128] = H[32x256] . W2^T + b2 ----
  {
    const int mtile  = wave & 1;
    const int ntbase = wave >> 1;
    FragU a[8];
    {
      const unsigned int* P = Hlds + (mtile * 16 + col) * HSTRIDE + hsel * 4;
      #pragma unroll
      for (int ks = 0; ks < 8; ++ks)
        #pragma unroll
        for (int v = 0; v < 4; ++v) {
          a[ks].u[v]     = P[ks * 16 + v];
          a[ks].u[4 + v] = P[ks * 16 + 8 + v];
        }
    }
    float* dst = out + (size_t)atom * N_NEIGH * OUT_F;
    #pragma unroll
    for (int t = 0; t < 2; ++t) {
      const int n = (ntbase + 4 * t) * 16 + col;
      const unsigned int* Q = Wlds + n * W2STRIDE + hsel * 4;
      v8f acc = {};
      #pragma unroll
      for (int ks = 0; ks < 8; ++ks) {
        FragU b;
        #pragma unroll
        for (int v = 0; v < 8; ++v) b.u[v] = Q[ks * 16 + (v & 3) + ((v >> 2) << 3)];
        acc = __builtin_amdgcn_wmma_f32_16x16x32_f16(
                  false, a[ks].h, false, b.h, (short)0, acc, false, false);
      }
      const float bias = b2g[s * OUT_F + n];
      #pragma unroll
      for (int r = 0; r < 8; ++r) {
        const int m = mtile * 16 + hsel * 8 + r;
        dst[m * OUT_F + n] = acc[r] + bias;
      }
    }
  }
}

extern "C" void kernel_launch(void* const* d_in, const int* in_sizes, int n_in,
                              void* d_out, int out_size, void* d_ws, size_t ws_size,
                              hipStream_t stream) {
  const float* features = (const float*)d_in[0];
  const int*   species  = (const int*)  d_in[1];
  const float* W1       = (const float*)d_in[2];
  const float* b1       = (const float*)d_in[3];
  const float* W2       = (const float*)d_in[4];
  const float* b2       = (const float*)d_in[5];
  float*       out      = (float*)d_out;
  (void)in_sizes; (void)n_in; (void)out_size;

  const size_t w1_elems = (size_t)N_SPECIES * HID * IN_F;   // 2M f16 each
  const size_t ws_need  = 2 * w1_elems * 2;                 // 8 MB (W1h + W2h)

  if (ws_size >= ws_need && d_ws != nullptr) {
    unsigned int* ws = (unsigned int*)d_ws;
    convert_weights_kernel<<<2048, 256, 0, stream>>>(W1, W2, ws);
    const unsigned int* W1h = ws;
    const unsigned int* W2h = ws + w1_elems / 2;            // +4 MB (u32 units)
    zcond_readout_kernel<true><<<N_ATOMS, 256, 0, stream>>>(
        features, species, W1, b1, W2, b2, W1h, W2h, out);
  } else {
    zcond_readout_kernel<false><<<N_ATOMS, 256, 0, stream>>>(
        features, species, W1, b1, W2, b2, nullptr, nullptr, out);
  }
}
